// Seq2Seq_80985903334085
// MI455X (gfx1250) — matrix operand order repaired and checked by reading, not compiled, not run on hardware
//
#include <hip/hip_runtime.h>
#include <hip/hip_bf16.h>

// ---------------------------------------------------------------------------
// Seq2seq LSTM + Bahdanau attention for MI455X (gfx1250), wave32 + WMMA.
//   V=32000, E=256, H=512, B=16, T=127
// Strategy:
//  * all GEMMs in f16 WMMA (v_wmma_f32_16x16x32_f16), f32 accumulate
//  * recurrences: one persistent 1024-thread workgroup (32 waves), state in
//    WMMA C-fragments; wave w owns hidden units [16w,16w+16)
//  * logits hoisted out of the recurrence into one large (2032,512,32000)
//    WMMA GEMM (dominant 66.6 GFLOP; W_proj f16 = 32MB lives in 192MB L2)
//  * LDS staging uses GLOBAL_LOAD_ASYNC_TO_LDS_B128 (ASYNCcnt path): no
//    VGPR bounce, direct memory->LDS DMA per lane.
// ---------------------------------------------------------------------------

#define T_SEQ 127
#define B_SZ  16
#define E_SZ  256
#define H_SZ  512
#define V_SZ  32000
#define NEG_BIG (-1e9f)

typedef __attribute__((ext_vector_type(16))) _Float16 v16h;
typedef __attribute__((ext_vector_type(8)))  float    v8f;

__device__ __forceinline__ float sigm_(float x) { return 1.0f / (1.0f + __expf(-x)); }

__device__ __forceinline__ v8f wmma16(v16h a, v16h b, v8f c) {
  // D = A(16x32 f16) * B(32x16 f16) + C(16x16 f32)
  return __builtin_amdgcn_wmma_f32_16x16x32_f16(false, a, false, b, (short)0, c,
                                                false, false);
}

// Async 16B copy: global (sgpr base + vgpr offset) -> LDS (vgpr byte offset).
// Tracked by ASYNCcnt; pair with ASYNC_WAIT() before the barrier.
#define ASYNC_CP16(ldsoff, src, goff)                                          \
  asm volatile("global_load_async_to_lds_b128 %0, %1, %2 offset:0"             \
               :: "v"(ldsoff), "v"(goff), "s"(src) : "memory")
#define ASYNC_WAIT() asm volatile("s_wait_asynccnt 0" ::: "memory")

// A fragment: 16x32 f16 tile from row-major [16][lda] at column kbase.
// A layout (matches B): lane L -> M=L&15, g=L>>4; element s -> K = 16g + s.
__device__ __forceinline__ v16h load_A_frag(const _Float16* base, int lda, int kbase) {
  int lane = threadIdx.x & 31;
  int m = lane & 15, g = lane >> 4;
  const _Float16* row = base + m * lda + kbase + 16 * g;
  v16h a;
#pragma unroll
  for (int i = 0; i < 16; ++i) a[i] = row[i];
  return a;
}

// B fragment: 32x16 f16 tile for output cols [nbase,nbase+16) at K=kbase.
// Weights pre-stored transposed Wt[n][k] (K contiguous) so each lane reads
// 16 contiguous f16 (two b128 loads).
// Layout: lane L -> N = nbase+(L&15), K half = 16*(L>>4); VGPR v -> K=2v(+half).
__device__ __forceinline__ v16h load_B_frag(const _Float16* Wt, int ldk, int nbase,
                                            int kbase) {
  int lane = threadIdx.x & 31;
  int n = nbase + (lane & 15), g = lane >> 4;
  const _Float16* col = Wt + (size_t)n * ldk + kbase + 16 * g;
  v16h b;
#pragma unroll
  for (int i = 0; i < 16; ++i) b[i] = col[i];
  return b;
}

// ---------------------------------------------------------------------------
// embedding gather + f16 cast:  out[t][b][e] = emb[tok[b][t]][e]
// ---------------------------------------------------------------------------
__global__ void embed_kernel(const int* __restrict__ tok, const float* __restrict__ emb,
                             _Float16* __restrict__ out) {
  int idx = blockIdx.x * blockDim.x + threadIdx.x;
  int total = T_SEQ * B_SZ * E_SZ;
  if (idx >= total) return;
  int e = idx & (E_SZ - 1);
  int b = (idx >> 8) & (B_SZ - 1);
  int t = idx >> 12;
  int token = tok[b * T_SEQ + t];
  out[idx] = (_Float16)emb[(size_t)token * E_SZ + e];
}

// transpose + cast: Wt[n][k] = (f16) W[k][n]   (W is [K][N] row-major f32)
__global__ void tcast_kernel(const float* __restrict__ W, _Float16* __restrict__ Wt,
                             int K, int N) {
  int idx = blockIdx.x * blockDim.x + threadIdx.x;
  if (idx >= K * N) return;
  int k = idx % K;
  int n = idx / K;
  Wt[idx] = (_Float16)W[(size_t)k * N + n];
}

// ---------------------------------------------------------------------------
// Encoder: persistent single-WG recurrence. 32 waves; wave w owns hidden
// units [16w,16w+16). c,h live in WMMA C-fragments.
// dyn LDS: xh[16][768] f16  (24.5KB) at LDS offset 0
// ---------------------------------------------------------------------------
__global__ void __launch_bounds__(1024) enc_kernel(
    const _Float16* __restrict__ x_seq,   // [T][16][256]
    const _Float16* __restrict__ Wt,      // [2048][768]  (Wenc transposed)
    const float* __restrict__ bias,       // [2048]
    const int* __restrict__ lens,         // [16]
    _Float16* __restrict__ out_h,         // [T][16][512]
    float* __restrict__ ch_out) {         // c[16][512] then h[16][512]
  extern __shared__ _Float16 xh[];        // [16][768]
  const int LDA = E_SZ + H_SZ;            // 768
  int tid = threadIdx.x, lane = tid & 31, wave = tid >> 5;
  int g = lane >> 4, nloc = lane & 15;
  int nb = wave * 16;
  int n = nb + nloc;

  float bi = bias[n], bj = bias[H_SZ + n], bf = bias[2 * H_SZ + n], bo = bias[3 * H_SZ + n];
  int lenr[8];
#pragma unroll
  for (int r = 0; r < 8; ++r) lenr[r] = lens[r + 8 * g];

  v8f c = {}, h = {};

  for (int t = 0; t < T_SEQ; ++t) {
    // stage x_t via async DMA: 8KB, 512 lanes x 16B; xh rows strided by LDA
    if (tid < 512) {
      unsigned m  = tid >> 5, e8 = tid & 31;          // chunk -> (row, 8xf16)
      unsigned goff = (unsigned)tid * 16;
      unsigned loff = m * (unsigned)(LDA * 2) + e8 * 16;
      const char* src = (const char*)(x_seq + (size_t)t * B_SZ * E_SZ);
      ASYNC_CP16(loff, src, goff);
    }
    // write current h into [x | h]
#pragma unroll
    for (int r = 0; r < 8; ++r) {
      int m = r + 8 * g;
      xh[m * LDA + E_SZ + n] = (_Float16)h[r];
    }
    ASYNC_WAIT();
    __syncthreads();

    v8f zi, zj, zf, zo;
#pragma unroll
    for (int r = 0; r < 8; ++r) { zi[r] = bi; zj[r] = bj; zf[r] = bf; zo[r] = bo; }

#pragma unroll 4
    for (int kk = 0; kk < LDA; kk += 32) {
      v16h a = load_A_frag(xh, LDA, kk);
      zi = wmma16(a, load_B_frag(Wt, LDA, nb, kk), zi);
      zj = wmma16(a, load_B_frag(Wt, LDA, H_SZ + nb, kk), zj);
      zf = wmma16(a, load_B_frag(Wt, LDA, 2 * H_SZ + nb, kk), zf);
      zo = wmma16(a, load_B_frag(Wt, LDA, 3 * H_SZ + nb, kk), zo);
    }
    __syncthreads();   // everyone done reading xh before next-step writes

#pragma unroll
    for (int r = 0; r < 8; ++r) {
      int m = r + 8 * g;
      float c2 = c[r] * sigm_(zf[r] + 1.0f) + sigm_(zi[r]) * tanhf(zj[r]);
      float h2 = tanhf(c2) * sigm_(zo[r]);
      bool valid = t < lenr[r];
      c[r] = valid ? c2 : c[r];
      h[r] = valid ? h2 : h[r];
      out_h[((size_t)t * B_SZ + m) * H_SZ + n] = (_Float16)(valid ? h2 : 0.0f);
    }
  }
#pragma unroll
  for (int r = 0; r < 8; ++r) {
    int m = r + 8 * g;
    ch_out[m * H_SZ + n] = c[r];
    ch_out[B_SZ * H_SZ + m * H_SZ + n] = h[r];
  }
}

// ---------------------------------------------------------------------------
// keys = enc_out @ W_mem : (2032, 512) x (512, 512) -> f32
// 8 waves/block, one 16x16 tile per wave.
// ---------------------------------------------------------------------------
__global__ void __launch_bounds__(256) keys_kernel(
    const _Float16* __restrict__ A,   // [2032][512]
    const _Float16* __restrict__ Bt,  // [512][512] (W_mem transposed)
    float* __restrict__ C) {          // [2032][512]
  int wid = (blockIdx.x * 256 + threadIdx.x) >> 5;
  int mt = wid >> 5;          // /32
  int nt = wid & 31;
  if (mt >= T_SEQ) return;
  int lane = threadIdx.x & 31, g = lane >> 4, nloc = lane & 15;
  v8f acc = {};
#pragma unroll 4
  for (int kk = 0; kk < H_SZ; kk += 32) {
    v16h a = load_A_frag(A + (size_t)mt * 16 * H_SZ, H_SZ, kk);
    acc = wmma16(a, load_B_frag(Bt, H_SZ, nt * 16, kk), acc);
  }
#pragma unroll
  for (int r = 0; r < 8; ++r) {
    int m = mt * 16 + r + 8 * g;
    int n = nt * 16 + nloc;
    C[(size_t)m * H_SZ + n] = acc[r];
  }
}

// ---------------------------------------------------------------------------
// Decoder: persistent single-WG recurrence with attention.
// dyn LDS carve (83,968 B), xh at LDS offset 0:
//   xh    [16][1280] f16  : [ emb(256) | attn(512) | h(512) ]
//   qbuf  [16][512]  f16
//   ctx   [16][512]  f16
//   score [16][128]  f32
//   av    [512]      f32
// ---------------------------------------------------------------------------
#define DEC_LDA 1280
__global__ void __launch_bounds__(1024) dec_kernel(
    const _Float16* __restrict__ x_seq,    // [T][16][256]
    const _Float16* __restrict__ Wd_t,     // [2048][1280]
    const float* __restrict__ bias,        // [2048]
    const _Float16* __restrict__ Wq_t,     // [512][512]
    const _Float16* __restrict__ Wa_t,     // [512][1024]
    const float* __restrict__ keys,        // [2032][512]
    const _Float16* __restrict__ enc_o,    // [2032][512]
    const float* __restrict__ attn_v,      // [512]
    const int* __restrict__ enc_len,       // [16]
    const float* __restrict__ ch_in,       // c/h from encoder
    _Float16* __restrict__ attn2_all) {    // [T][16][512]
  extern __shared__ char smem[];
  _Float16* xh   = (_Float16*)smem;                 // 16*1280
  _Float16* qbuf = xh + 16 * DEC_LDA;               // 16*512
  _Float16* ctxb = qbuf + 16 * H_SZ;                // 16*512
  float*    sc   = (float*)(ctxb + 16 * H_SZ);      // 16*128
  float*    av   = sc + 16 * 128;                   // 512

  int tid = threadIdx.x, lane = tid & 31, wave = tid >> 5;
  int g = lane >> 4, nloc = lane & 15;
  int nb = wave * 16;
  int n = nb + nloc;

  float bi = bias[n], bj = bias[H_SZ + n], bf = bias[2 * H_SZ + n], bo = bias[3 * H_SZ + n];

  v8f c, h, attn = {};
#pragma unroll
  for (int r = 0; r < 8; ++r) {
    int m = r + 8 * g;
    c[r] = ch_in[m * H_SZ + n];
    h[r] = ch_in[B_SZ * H_SZ + m * H_SZ + n];
  }
  for (int i = tid; i < H_SZ; i += 1024) av[i] = attn_v[i];
  __syncthreads();

  for (int t = 0; t < T_SEQ; ++t) {
    // ---- stage x_t via async DMA (xh rows strided by DEC_LDA) ----
    if (tid < 512) {
      unsigned m  = tid >> 5, e8 = tid & 31;
      unsigned goff = (unsigned)tid * 16;
      unsigned loff = m * (unsigned)(DEC_LDA * 2) + e8 * 16;
      const char* src = (const char*)(x_seq + (size_t)t * B_SZ * E_SZ);
      ASYNC_CP16(loff, src, goff);
    }
    // ---- stage attn and h carries ----
#pragma unroll
    for (int r = 0; r < 8; ++r) {
      int m = r + 8 * g;
      xh[m * DEC_LDA + E_SZ + n] = (_Float16)attn[r];
      xh[m * DEC_LDA + E_SZ + H_SZ + n] = (_Float16)h[r];
    }
    ASYNC_WAIT();
    __syncthreads();

    // ---- LSTM gates: z = [xi;h](16x1280) @ Wdec ----
    v8f zi, zj, zf, zo;
#pragma unroll
    for (int r = 0; r < 8; ++r) { zi[r] = bi; zj[r] = bj; zf[r] = bf; zo[r] = bo; }
#pragma unroll 4
    for (int kk = 0; kk < DEC_LDA; kk += 32) {
      v16h a = load_A_frag(xh, DEC_LDA, kk);
      zi = wmma16(a, load_B_frag(Wd_t, DEC_LDA, nb, kk), zi);
      zj = wmma16(a, load_B_frag(Wd_t, DEC_LDA, H_SZ + nb, kk), zj);
      zf = wmma16(a, load_B_frag(Wd_t, DEC_LDA, 2 * H_SZ + nb, kk), zf);
      zo = wmma16(a, load_B_frag(Wd_t, DEC_LDA, 3 * H_SZ + nb, kk), zo);
    }
    __syncthreads();
#pragma unroll
    for (int r = 0; r < 8; ++r) {
      float c2 = c[r] * sigm_(zf[r] + 1.0f) + sigm_(zi[r]) * tanhf(zj[r]);
      float h2 = tanhf(c2) * sigm_(zo[r]);
      c[r] = c2;
      h[r] = h2;
    }
    // publish h2 (overwrite h slot of xh)
#pragma unroll
    for (int r = 0; r < 8; ++r) {
      int m = r + 8 * g;
      xh[m * DEC_LDA + E_SZ + H_SZ + n] = (_Float16)h[r];
    }
    __syncthreads();

    // ---- q = h2 @ W_query (16x512x512) ----
    v8f q = {};
#pragma unroll 4
    for (int kk = 0; kk < H_SZ; kk += 32) {
      v16h a = load_A_frag(xh + E_SZ + H_SZ, DEC_LDA, kk);
      q = wmma16(a, load_B_frag(Wq_t, H_SZ, nb, kk), q);
    }
#pragma unroll
    for (int r = 0; r < 8; ++r) {
      int m = r + 8 * g;
      qbuf[m * H_SZ + n] = (_Float16)q[r];
    }
    __syncthreads();

    // ---- scores: one (b,tt) pair per wave iteration, lanes over H ----
    for (int p = wave; p < T_SEQ * B_SZ; p += 32) {
      int tt = p >> 4, b = p & 15;
      float s = 0.0f;
      for (int hh = lane; hh < H_SZ; hh += 32)
        s += tanhf(keys[(size_t)p * H_SZ + hh] + (float)qbuf[b * H_SZ + hh]) * av[hh];
#pragma unroll
      for (int off = 16; off > 0; off >>= 1) s += __shfl_xor(s, off, 32);
      if (lane == 0)
        sc[b * 128 + tt] = (tt < enc_len[b]) ? s : NEG_BIG;
    }
    __syncthreads();

    // ---- softmax over tt (wave w handles row b=w) ----
    if (wave < B_SZ) {
      int b = wave;
      float v[4], mx = NEG_BIG;
#pragma unroll
      for (int i = 0; i < 4; ++i) {
        int tt = lane + 32 * i;
        v[i] = (tt < T_SEQ) ? sc[b * 128 + tt] : NEG_BIG;
        mx = fmaxf(mx, v[i]);
      }
#pragma unroll
      for (int off = 16; off > 0; off >>= 1) mx = fmaxf(mx, __shfl_xor(mx, off, 32));
      float sum = 0.0f;
#pragma unroll
      for (int i = 0; i < 4; ++i) {
        int tt = lane + 32 * i;
        v[i] = (tt < T_SEQ) ? __expf(v[i] - mx) : 0.0f;
        sum += v[i];
      }
#pragma unroll
      for (int off = 16; off > 0; off >>= 1) sum += __shfl_xor(sum, off, 32);
      float inv = 1.0f / sum;
#pragma unroll
      for (int i = 0; i < 4; ++i) {
        int tt = lane + 32 * i;
        if (tt < T_SEQ) sc[b * 128 + tt] = v[i] * inv;
      }
    }
    __syncthreads();

    // ---- ctx[b][h] = sum_tt align * enc_out ----
    for (int i = tid; i < B_SZ * H_SZ; i += 1024) {
      int b = i >> 9, hh = i & 511;
      float acc = 0.0f;
      for (int tt = 0; tt < T_SEQ; ++tt)
        acc += sc[b * 128 + tt] * (float)enc_o[((size_t)tt * B_SZ + b) * H_SZ + hh];
      ctxb[b * H_SZ + hh] = (_Float16)acc;
    }
    __syncthreads();

    // ---- attn2 = [h2; ctx](16x1024) @ W_attn ----
    v8f a2 = {};
#pragma unroll 4
    for (int kk = 0; kk < H_SZ; kk += 32) {
      v16h a = load_A_frag(xh + E_SZ + H_SZ, DEC_LDA, kk);
      a2 = wmma16(a, load_B_frag(Wa_t, 2 * H_SZ, nb, kk), a2);
    }
#pragma unroll 4
    for (int kk = 0; kk < H_SZ; kk += 32) {
      v16h a = load_A_frag(ctxb, H_SZ, kk);
      a2 = wmma16(a, load_B_frag(Wa_t, 2 * H_SZ, nb, H_SZ + kk), a2);
    }
    attn = a2;
#pragma unroll
    for (int r = 0; r < 8; ++r) {
      int m = r + 8 * g;
      attn2_all[((size_t)t * B_SZ + m) * H_SZ + n] = (_Float16)a2[r];
    }
    __syncthreads();  // xh/ctxb reads done before next-step writes
  }
}

// ---------------------------------------------------------------------------
// logits = attn2_all @ W_proj : (2032, 512) x (512, 32000), masked + [B,T,V]
// Block: 8 waves x 64 cols = 512 cols, one 16-row panel async-staged in LDS.
// ---------------------------------------------------------------------------
__global__ void __launch_bounds__(256) logits_kernel(
    const _Float16* __restrict__ A,     // [2032][512]
    const _Float16* __restrict__ Wp_t,  // [32000][512]
    const int* __restrict__ dec_len,    // [16]
    float* __restrict__ out) {          // [16][127][32000]
  extern __shared__ _Float16 As[];      // [16][512] at LDS offset 0
  int mt = blockIdx.x;                  // 0..126
  int tid = threadIdx.x, lane = tid & 31, wave = tid >> 5;
  int g = lane >> 4, nloc = lane & 15;

  // async-stage 16KB A panel: 256 lanes x 16B x 4
  {
    const char* src = (const char*)(A + (size_t)mt * 16 * H_SZ);
#pragma unroll
    for (int it = 0; it < 4; ++it) {
      unsigned off16 = ((unsigned)tid + it * 256u) * 16u;
      ASYNC_CP16(off16, src, off16);   // contiguous panel: lds off == glb off
    }
    ASYNC_WAIT();
  }
  __syncthreads();

  int nwbase = blockIdx.y * 512 + wave * 64;
  v8f acc0 = {}, acc1 = {}, acc2 = {}, acc3 = {};
#pragma unroll 2
  for (int kk = 0; kk < H_SZ; kk += 32) {
    v16h a = load_A_frag(As, H_SZ, kk);
    if (nwbase < V_SZ)      acc0 = wmma16(a, load_B_frag(Wp_t, H_SZ, nwbase, kk), acc0);
    if (nwbase + 16 < V_SZ) acc1 = wmma16(a, load_B_frag(Wp_t, H_SZ, nwbase + 16, kk), acc1);
    if (nwbase + 32 < V_SZ) acc2 = wmma16(a, load_B_frag(Wp_t, H_SZ, nwbase + 32, kk), acc2);
    if (nwbase + 48 < V_SZ) acc3 = wmma16(a, load_B_frag(Wp_t, H_SZ, nwbase + 48, kk), acc3);
    // keep the W_proj stream warm in L2/WGP$ (emits global_prefetch_b8)
    __builtin_prefetch(Wp_t + (size_t)(nwbase + (lane & 15)) * H_SZ + kk + 64, 0, 1);
  }

  int t = mt;
#pragma unroll
  for (int i = 0; i < 4; ++i) {
    int n0 = nwbase + 16 * i;
    if (n0 >= V_SZ) break;
    v8f acc = (i == 0) ? acc0 : (i == 1) ? acc1 : (i == 2) ? acc2 : acc3;
#pragma unroll
    for (int r = 0; r < 8; ++r) {
      int b = r + 8 * g;               // row m -> batch index (A is [t][b])
      float v = (t < dec_len[b]) ? acc[r] : 0.0f;
      out[((size_t)b * T_SEQ + t) * V_SZ + n0 + nloc] = v;
    }
  }
}

// ---------------------------------------------------------------------------
// Host launcher
// ---------------------------------------------------------------------------
extern "C" void kernel_launch(void* const* d_in, const int* in_sizes, int n_in,
                              void* d_out, int out_size, void* d_ws, size_t ws_size,
                              hipStream_t stream) {
  const int*   enc_tok = (const int*)d_in[0];
  const int*   dec_tok = (const int*)d_in[1];
  const int*   enc_len = (const int*)d_in[2];
  const int*   dec_len = (const int*)d_in[3];
  const float* emb     = (const float*)d_in[4];
  const float* W_enc   = (const float*)d_in[5];
  const float* b_enc   = (const float*)d_in[6];
  const float* W_dec   = (const float*)d_in[7];
  const float* b_dec   = (const float*)d_in[8];
  const float* W_mem   = (const float*)d_in[9];
  const float* W_query = (const float*)d_in[10];
  const float* attn_v  = (const float*)d_in[11];
  const float* W_attn  = (const float*)d_in[12];
  const float* W_proj  = (const float*)d_in[13];
  float* out = (float*)d_out;

  char* ws = (char*)d_ws;
  size_t off = 0;
  auto carve = [&](size_t bytes) -> void* {
    void* p = ws + off;
    off = (off + bytes + 255) & ~(size_t)255;
    return p;
  };
  _Float16* enc_x   = (_Float16*)carve((size_t)T_SEQ * B_SZ * E_SZ * 2);
  _Float16* dec_x   = (_Float16*)carve((size_t)T_SEQ * B_SZ * E_SZ * 2);
  _Float16* Wenc_t  = (_Float16*)carve((size_t)4 * H_SZ * (E_SZ + H_SZ) * 2);
  _Float16* Wdec_t  = (_Float16*)carve((size_t)4 * H_SZ * (E_SZ + 2 * H_SZ) * 2);
  _Float16* Wmem_t  = (_Float16*)carve((size_t)H_SZ * H_SZ * 2);
  _Float16* Wq_t    = (_Float16*)carve((size_t)H_SZ * H_SZ * 2);
  _Float16* Wattn_t = (_Float16*)carve((size_t)H_SZ * 2 * H_SZ * 2);
  _Float16* Wproj_t = (_Float16*)carve((size_t)V_SZ * H_SZ * 2);
  _Float16* enc_out = (_Float16*)carve((size_t)T_SEQ * B_SZ * H_SZ * 2);
  float*    keys    = (float*)   carve((size_t)T_SEQ * B_SZ * H_SZ * 4);
  _Float16* attn2   = (_Float16*)carve((size_t)T_SEQ * B_SZ * H_SZ * 2);
  float*    ch      = (float*)   carve((size_t)2 * B_SZ * H_SZ * 4);

  const int EMB_N = T_SEQ * B_SZ * E_SZ;
  embed_kernel<<<(EMB_N + 255) / 256, 256, 0, stream>>>(enc_tok, emb, enc_x);
  embed_kernel<<<(EMB_N + 255) / 256, 256, 0, stream>>>(dec_tok, emb, dec_x);

  auto tc = [&](const float* W, _Float16* Wt, int K, int N) {
    int tot = K * N;
    tcast_kernel<<<(tot + 255) / 256, 256, 0, stream>>>(W, Wt, K, N);
  };
  tc(W_enc,   Wenc_t,  E_SZ + H_SZ,     4 * H_SZ);
  tc(W_dec,   Wdec_t,  E_SZ + 2 * H_SZ, 4 * H_SZ);
  tc(W_mem,   Wmem_t,  H_SZ,            H_SZ);
  tc(W_query, Wq_t,    H_SZ,            H_SZ);
  tc(W_attn,  Wattn_t, 2 * H_SZ,        H_SZ);
  tc(W_proj,  Wproj_t, H_SZ,            V_SZ);

  // encoder recurrence (single persistent workgroup)
  enc_kernel<<<1, 1024, (size_t)B_SZ * (E_SZ + H_SZ) * 2, stream>>>(
      enc_x, Wenc_t, b_enc, enc_len, enc_out, ch);

  // keys = enc_out @ W_mem
  keys_kernel<<<(T_SEQ * 32) / 8, 256, 0, stream>>>(enc_out, Wmem_t, keys);

  // decoder recurrence (single persistent workgroup)
  size_t dec_lds = (size_t)B_SZ * DEC_LDA * 2   // xh
                 + (size_t)B_SZ * H_SZ * 2      // qbuf
                 + (size_t)B_SZ * H_SZ * 2      // ctx
                 + (size_t)B_SZ * 128 * 4       // score
                 + (size_t)H_SZ * 4;            // attn_v
  dec_kernel<<<1, 1024, dec_lds, stream>>>(dec_x, Wdec_t, b_dec, Wq_t, Wattn_t,
                                           keys, enc_out, attn_v, enc_len, ch, attn2);

  // logits = attn2 @ W_proj  (dominant GEMM)
  dim3 lg(T_SEQ, (V_SZ + 511) / 512);
  logits_kernel<<<lg, 256, (size_t)B_SZ * H_SZ * 2, stream>>>(attn2, Wproj_t,
                                                              dec_len, out);
}
